// PointConv_72662256714582
// MI455X (gfx1250) — matrix operand order repaired and checked by reading, not compiled, not run on hardware
//
#include <hip/hip_runtime.h>

// PointConv for MI455X (gfx1250):
//  - MLP1 (rel_pos @ W1 + b1):  V_WMMA_F32_16X16X4_F32  (exact fp32, bias folded in)
//  - MLP2 (h @ W2 + b2):        V_WMMA_F32_16X16X32_F16 (f32 accumulate, b2 in C-init)
//  - x gather: global_load_async_to_lds_b128 (ASYNCcnt path, no VGPR round trip)
// One wave32 per point; gathers served from the 192MB L2 (x = 16MB resident).

typedef __attribute__((ext_vector_type(16))) _Float16 v16h;
typedef __attribute__((ext_vector_type(8)))  _Float16 v8h;
typedef __attribute__((ext_vector_type(8)))  float    v8f;
typedef __attribute__((ext_vector_type(2)))  float    v2f;

#define BB 4
#define NN 16384
#define KK 32          // neighbors == wave32 lanes
#define DD 64          // feature dim
#define WAVES_PER_BLOCK 4
#define THREADS (WAVES_PER_BLOCK * 32)

__global__ __launch_bounds__(THREADS)
void pointconv_wmma_kernel(const float* __restrict__ x,
                           const float* __restrict__ pos,
                           const int*   __restrict__ nbr,
                           const float* __restrict__ W1,
                           const float* __restrict__ b1,
                           const float* __restrict__ W2,
                           const float* __restrict__ b2,
                           float* __restrict__ out)
{
    __shared__ __align__(32) _Float16 sW2T[DD * DD];                // W2^T f16, 8KB
    __shared__ __align__(16) _Float16 sH[WAVES_PER_BLOCK][KK * DD]; // 4KB/wave
    __shared__ __align__(16) float    sX[WAVES_PER_BLOCK][KK * DD]; // 8KB/wave

    const int tid  = threadIdx.x;
    const int lane = tid & 31;
    const int wv   = tid >> 5;
    const int n16  = lane & 15;
    const int hi   = lane >> 4;

    // ---- stage W2^T (f16) once per block ----
    for (int i = tid; i < DD * DD; i += THREADS) {
        const int g = i >> 6, f = i & 63;
        sW2T[f * DD + g] = (_Float16)W2[i];
    }
    __syncthreads();

    // ---- hoist GEMM2 B fragments (W2): lanes 0-15 hold K=0..15, 16-31 K=16..31 ----
    v16h bf[2][4];
    #pragma unroll
    for (int Kt = 0; Kt < 2; ++Kt)
        #pragma unroll
        for (int Nt = 0; Nt < 4; ++Nt)
            bf[Kt][Nt] = *(const v16h*)&sW2T[(16 * Nt + n16) * DD + 32 * Kt + 16 * hi];

    // ---- hoist GEMM1 B fragments (W1 rows 0..2, row 3 = b1) : K = 2*hi + e ----
    v2f bW1[4];
    #pragma unroll
    for (int Nt = 0; Nt < 4; ++Nt) {
        #pragma unroll
        for (int e = 0; e < 2; ++e) {
            const int d = 2 * hi + e;
            bW1[Nt][e] = (d < 3) ? W1[d * DD + 16 * Nt + n16] : b1[16 * Nt + n16];
        }
    }

    float b2v[4];
    #pragma unroll
    for (int Nt = 0; Nt < 4; ++Nt) b2v[Nt] = b2[16 * Nt + n16];

    // LDS byte offset for async DMA destination (low 32 bits of flat LDS addr)
    const unsigned ldsX = (unsigned)(uintptr_t)&sX[wv][lane * DD];

    const int totalWaves = (int)gridDim.x * WAVES_PER_BLOCK;
    const int wgid       = (int)blockIdx.x * WAVES_PER_BLOCK + wv;

    for (int pt = wgid; pt < BB * NN; pt += totalWaves) {     // wave-uniform loop
        const int b = pt >> 14;                               // pt / NN
        const float cx = pos[pt * 3 + 0];
        const float cy = pos[pt * 3 + 1];
        const float cz = pos[pt * 3 + 2];

        // lane k owns neighbor k
        const int idx = nbr[pt * KK + lane];
        const int nb  = b * NN + idx;
        const float rx = cx - pos[nb * 3 + 0];
        const float ry = cy - pos[nb * 3 + 1];
        const float rz = cz - pos[nb * 3 + 2];

        // ---- async DMA: x[idx_k][0:64] -> LDS (64B per lane, 4x b128) ----
        {
            const unsigned long long ga =
                (unsigned long long)(uintptr_t)&x[(size_t)nb * DD];
            asm volatile(
                "global_load_async_to_lds_b128 %0, %1, off\n\t"
                "global_load_async_to_lds_b128 %0, %1, off offset:16\n\t"
                "global_load_async_to_lds_b128 %0, %1, off offset:32\n\t"
                "global_load_async_to_lds_b128 %0, %1, off offset:48"
                :: "v"(ldsX), "v"(ga) : "memory");
        }

        // ---- GEMM1 on WMMA: h_pre = [rel,1] @ [W1;b1]  (16x16x4 f32) ----
        // 32-bit A 16x4 layout: lane m (0-15) VGPR e -> K = 2*hi + e
        #pragma unroll
        for (int Mt = 0; Mt < 2; ++Mt) {
            const float sx = __shfl(rx, 16 * Mt + n16, 32);
            const float sy = __shfl(ry, 16 * Mt + n16, 32);
            const float sz = __shfl(rz, 16 * Mt + n16, 32);
            v2f a1;
            a1[0] = hi ? sz : sx;
            a1[1] = hi ? 1.0f : sy;
            #pragma unroll
            for (int Nt = 0; Nt < 4; ++Nt) {
                v8f hc;
                #pragma unroll
                for (int r = 0; r < 8; ++r) hc[r] = 0.f;
                hc = __builtin_amdgcn_wmma_f32_16x16x4_f32(
                        false, a1, false, bW1[Nt], (short)0, hc, false, false);
                // LeakyReLU(0.1) == max(x, 0.1x) for slope in (0,1): v_mul + v_max
                #pragma unroll
                for (int r = 0; r < 8; ++r) {
                    const float v = fmaxf(hc[r], 0.1f * hc[r]);
                    sH[wv][(16 * Mt + 8 * hi + r) * DD + 16 * Nt + n16] = (_Float16)v;
                }
            }
        }

        // ---- GEMM2 A fragments (h): k = 32*Kt + 16*hi + s, same order as B (DS in-order w/ stores) ----
        v16h af[2][2];
        #pragma unroll
        for (int Mt = 0; Mt < 2; ++Mt)
            #pragma unroll
            for (int Kt = 0; Kt < 2; ++Kt) {
                const _Float16* hb = &sH[wv][(16 * Mt + n16) * DD + 32 * Kt + 16 * hi];
                v8h lo = *(const v8h*)&hb[0];
                v8h hh = *(const v8h*)&hb[8];
                af[Mt][Kt] = __builtin_shufflevector(lo, hh,
                    0, 1, 2, 3, 4, 5, 6, 7, 8, 9, 10, 11, 12, 13, 14, 15);
            }

        // x DMA must have landed before the fused reduction reads sX
        asm volatile("s_wait_asynccnt 0x0" ::: "memory");

        // ---- GEMM2 (16x16x32 f16, f32 acc) + fused weighted reduction ----
        float outp[4] = {0.f, 0.f, 0.f, 0.f};
        #pragma unroll
        for (int Mt = 0; Mt < 2; ++Mt) {
            #pragma unroll
            for (int Nt = 0; Nt < 4; ++Nt) {
                v8f c;
                #pragma unroll
                for (int r = 0; r < 8; ++r) c[r] = b2v[Nt];   // fold b2 into C
                c = __builtin_amdgcn_wmma_f32_16x16x32_f16(
                        false, af[Mt][0], false, bf[0][Nt], (short)0, c, false, false);
                c = __builtin_amdgcn_wmma_f32_16x16x32_f16(
                        false, af[Mt][1], false, bf[1][Nt], (short)0, c, false, false);
                // D layout: VGPR r -> k = 16*Mt + 8*hi + r, f = 16*Nt + n16
                float s = 0.f;
                #pragma unroll
                for (int r = 0; r < 8; ++r)
                    s += c[r] * sX[wv][(16 * Mt + 8 * hi + r) * DD + 16 * Nt + n16];
                outp[Nt] += s;
            }
        }

        // combine the two k-halves (lane L <-> L^16 share f) and store
        float* orow = &out[(size_t)pt * DD];
        #pragma unroll
        for (int Nt = 0; Nt < 4; ++Nt) {
            float tot = outp[Nt] + __shfl_xor(outp[Nt], 16, 32);
            if (hi == 0) orow[16 * Nt + n16] = tot;
        }
    }
}

extern "C" void kernel_launch(void* const* d_in, const int* in_sizes, int n_in,
                              void* d_out, int out_size, void* d_ws, size_t ws_size,
                              hipStream_t stream) {
    const float* x   = (const float*)d_in[0];
    const float* pos = (const float*)d_in[1];
    const int*   nbr = (const int*)d_in[2];
    const float* W1  = (const float*)d_in[3];
    const float* b1  = (const float*)d_in[4];
    const float* W2  = (const float*)d_in[5];
    const float* b2  = (const float*)d_in[6];
    float* out = (float*)d_out;

    dim3 grid(2048), block(THREADS);   // 8192 waves, 8 points/wave grid-stride
    pointconv_wmma_kernel<<<grid, block, 0, stream>>>(x, pos, nbr, W1, b1, W2, b2, out);
}
